// E8ResidualBottleneck_43224550867044
// MI455X (gfx1250) — matrix operand order repaired and checked by reading, not compiled, not run on hardware
//
#include <hip/hip_runtime.h>

typedef unsigned int   u32;
typedef unsigned short u16;

typedef __attribute__((ext_vector_type(16))) __bf16 v16bf;
typedef __attribute__((ext_vector_type(8)))  float  v8f;

struct alignas(16) U128 { u32 x, y, z, w; };

union U16bf {
  v16bf v;
  U128  q[2];
  u32   u[8];
};

#define BULK   2048
#define TOWER  512
#define NLEV   8
#define MTILE  64
#define TS2_STRIDE 520   // u16 elements; 1040B rows, 16B aligned

__device__ __forceinline__ u16 f2bf(float f) {
  u32 u = __float_as_uint(f);
  u += 0x7FFFu + ((u >> 16) & 1u);   // RNE
  return (u16)(u >> 16);
}

#if __has_builtin(__builtin_amdgcn_cvt_pk_bf16_f32)
typedef __attribute__((ext_vector_type(2))) __bf16 v2bf;
__device__ __forceinline__ u32 pk2(float a, float b) {   // a -> low half, b -> high half
  union { v2bf v; u32 u; } c;
  c.v = __builtin_amdgcn_cvt_pk_bf16_f32(a, b);
  return c.u;
}
#else
__device__ __forceinline__ u32 pk2(float a, float b) {
  union { __bf16 h[2]; u32 u; } c;
  c.h[0] = (__bf16)a;   // let the backend pick a native cvt if available
  c.h[1] = (__bf16)b;
  return c.u;
}
#endif

__device__ __forceinline__ float bf2f(u16 h) {
  return __uint_as_float(((u32)h) << 16);
}
__device__ __forceinline__ v8f wmma_bf16(const U16bf& a, const U16bf& b, v8f c) {
  return __builtin_amdgcn_wmma_f32_16x16x32_bf16(false, a.v, false, b.v, (short)0, c, false, false);
}

// nearest point of D8 lattice (even-parity integer vectors), no dynamic indexing
__device__ __forceinline__ void nd8(const float x[8], float f[8]) {
  int s = 0; float mx = -1.0f; int mi = 0; float me = 0.0f;
  #pragma unroll
  for (int i = 0; i < 8; ++i) {
    f[i] = rintf(x[i]);
    s += (int)f[i];
    float e = x[i] - f[i];
    float ae = fabsf(e);
    if (ae > mx) { mx = ae; mi = i; me = e; }
  }
  if (s & 1) {
    float adj = (me >= 0.0f) ? 1.0f : -1.0f;
    #pragma unroll
    for (int i = 0; i < 8; ++i) if (i == mi) f[i] += adj;
  }
}

// ---- prep: bf16-convert + transpose weights into ws ----
// wbtT[n][k] 512x2048, wtbT[n][k] 2048x512, wt8T[n][k] 16x512 (cols 8..15 zero)
__global__ __launch_bounds__(256) void e8_prep(const float* __restrict__ Wbt,
                                               const float* __restrict__ Wtb,
                                               const float* __restrict__ Wt8,
                                               u16* __restrict__ wbtT,
                                               u16* __restrict__ wtbT,
                                               u16* __restrict__ wt8T) {
  int i = blockIdx.x * 256 + threadIdx.x;
  const int N1 = BULK * TOWER;
  if (i < N1) {                              // W_bt[k][n] (2048x512) -> wbtT[n][k]
    int n = i >> 11, k = i & 2047;
    wbtT[i] = f2bf(Wbt[(size_t)k * TOWER + n]);
  } else if (i < 2 * N1) {                   // W_tb[k][n] (512x2048) -> wtbT[n][k]
    int j = i - N1;
    int n = j >> 9, k = j & 511;
    wtbT[j] = f2bf(Wtb[(size_t)k * BULK + n]);
  } else if (i < 2 * N1 + 16 * TOWER) {      // W_t8[k][n] (512x8) -> wt8T[n][k], zero-padded
    int j = i - 2 * N1;
    int n = j >> 9, k = j & 511;
    wt8T[j] = (n < 8) ? f2bf(Wt8[(size_t)k * 8 + n]) : (u16)0;
  }
}

// ---- analytic LN stats for the 8->512 expansion:
// sb[0..7]=sum_n W_8t[j][n]; sb[8..15]=sum_n b_8t[n]*W_8t[j][n];
// sb[16..79]=Gram G[j][k]; sb[80]=sum b; sb[81]=sum b^2
__global__ void e8_stats(const float* __restrict__ W_8t,
                         const float* __restrict__ b_8t,
                         float* __restrict__ sb) {
  int t = threadIdx.x;   // 128 threads
  if (t < 8) {
    float sw = 0.f, swb = 0.f;
    for (int n = 0; n < TOWER; ++n) { float w = W_8t[t * TOWER + n]; sw += w; swb += b_8t[n] * w; }
    sb[t] = sw; sb[8 + t] = swb;
  } else if (t < 72) {
    int j = (t - 8) >> 3, k = (t - 8) & 7;
    float g = 0.f;
    for (int n = 0; n < TOWER; ++n) g += W_8t[j * TOWER + n] * W_8t[k * TOWER + n];
    sb[16 + (t - 8)] = g;
  } else if (t == 72) {
    float s = 0.f, s2 = 0.f;
    for (int n = 0; n < TOWER; ++n) { float b = b_8t[n]; s += b; s2 += b * b; }
    sb[80] = s; sb[81] = s2;
  }
}

// ================= encode: x -> tower -> e8 -> quantize -> t2(bf16) =================
__global__ __launch_bounds__(256) void e8_encode(
    const float* __restrict__ x,
    const u16*  __restrict__ wbtT,
    const u16*  __restrict__ wt8T,
    const float* __restrict__ b_bt,
    const float* __restrict__ ln_et_g, const float* __restrict__ ln_et_b,
    const float* __restrict__ b_t8,
    const float* __restrict__ rms_g,
    const float* __restrict__ W_8t,    const float* __restrict__ b_8t,
    const float* __restrict__ ln_dt_g, const float* __restrict__ ln_dt_b,
    const float* __restrict__ sb,
    u16* __restrict__ t2g) {

  __shared__ __align__(16) u16 Ts2[MTILE][TS2_STRIDE];  // normalized tower, bf16
  __shared__ float Tb_bbt[TOWER], Tg_et[TOWER], Tb_et[TOWER];
  __shared__ float Tb_8t[TOWER],  Tg_dt[TOWER], Tb_dt[TOWER];
  __shared__ float E8s[MTILE][8], Qs[MTILE][8];
  __shared__ float RowMV[MTILE][2];
  __shared__ float RowStats[2 * MTILE];

  const int t  = threadIdx.x;
  const int w  = t >> 5;
  const int L  = t & 31;
  const int lo = L & 15;
  const int hi = L >> 4;
  const int mg = w >> 1;        // 0..3 : which 16-row group
  const int ng = w & 1;         // 0..1 : which 256-col half
  const int row0 = blockIdx.x * MTILE;

  for (int i = t; i < TOWER; i += 256) {
    Tb_bbt[i] = b_bt[i];
    Tg_et[i]  = ln_et_g[i]; Tb_et[i] = ln_et_b[i];
    Tb_8t[i]  = b_8t[i];
    Tg_dt[i]  = ln_dt_g[i]; Tb_dt[i] = ln_dt_b[i];
  }
  if (t < 2 * MTILE) RowStats[t] = 0.0f;
  __syncthreads();

  // ---------------- GEMM1: x(64x2048) @ W_bt -> tower(64x512) ----------------
  v8f c1[16] = {};
  {
    const float* xr0 = x + (size_t)(row0 + mg * 16 + lo) * BULK + hi * 8;
    for (int k0 = 0; k0 < BULK; k0 += 32) {
      const float* xr = xr0 + k0;
      float4 f0 = *(const float4*)(xr);
      float4 f1 = *(const float4*)(xr + 4);
      float4 f2 = *(const float4*)(xr + 16);
      float4 f3 = *(const float4*)(xr + 20);
      U16bf a;
      a.u[0] = pk2(f0.x, f0.y); a.u[1] = pk2(f0.z, f0.w);
      a.u[2] = pk2(f1.x, f1.y); a.u[3] = pk2(f1.z, f1.w);
      a.u[4] = pk2(f2.x, f2.y); a.u[5] = pk2(f2.z, f2.w);
      a.u[6] = pk2(f3.x, f3.y); a.u[7] = pk2(f3.z, f3.w);
      #pragma unroll
      for (int f = 0; f < 16; ++f) {
        int n = ng * 256 + f * 16 + lo;
        const u16* wr = wbtT + (size_t)n * BULK + k0 + hi * 8;
        U16bf b;
        b.q[0] = *(const U128*)(wr);
        b.q[1] = *(const U128*)(wr + 16);
        c1[f] = wmma_bf16(a, b, c1[f]);
      }
    }
  }
  // ---- encode LN directly from accumulators ----
  {
    float s[8], s2[8];
    #pragma unroll
    for (int v = 0; v < 8; ++v) { s[v] = 0.f; s2[v] = 0.f; }
    #pragma unroll
    for (int f = 0; f < 16; ++f) {
      float bias = Tb_bbt[ng * 256 + f * 16 + lo];
      #pragma unroll
      for (int v = 0; v < 8; ++v) {
        float z = c1[f][v] + bias;
        c1[f][v] = z; s[v] += z; s2[v] += z * z;
      }
    }
    #pragma unroll
    for (int m = 1; m < 16; m <<= 1) {
      #pragma unroll
      for (int v = 0; v < 8; ++v) {
        s[v]  += __shfl_xor(s[v],  m, 32);
        s2[v] += __shfl_xor(s2[v], m, 32);
      }
    }
    if (lo == 0) {
      #pragma unroll
      for (int v = 0; v < 8; ++v) {
        int rl = mg * 16 + hi * 8 + v;
        atomicAdd(&RowStats[rl * 2 + 0], s[v]);
        atomicAdd(&RowStats[rl * 2 + 1], s2[v]);
      }
    }
  }
  __syncthreads();
  {
    float mu[8], rs[8];
    #pragma unroll
    for (int v = 0; v < 8; ++v) {
      int rl = mg * 16 + hi * 8 + v;
      float m_  = RowStats[rl * 2 + 0] * (1.0f / 512.0f);
      float var = RowStats[rl * 2 + 1] * (1.0f / 512.0f) - m_ * m_;
      mu[v] = m_; rs[v] = rsqrtf(var + 1e-6f);
    }
    #pragma unroll
    for (int f = 0; f < 16; ++f) {
      int col = ng * 256 + f * 16 + lo;
      float g = Tg_et[col], b = Tb_et[col];
      #pragma unroll
      for (int v = 0; v < 8; ++v) {
        float y = (c1[f][v] - mu[v]) * rs[v] * g + b;
        Ts2[mg * 16 + hi * 8 + v][col] = f2bf(y);
      }
    }
  }
  __syncthreads();

  // ------- GEMM2 via WMMA: tower(64x512) @ W_t8(512x16-padded) -> e8 -------
  if (w < 4) {                   // wave w handles 16-row group w, full K=512
    v8f ce = {};
    for (int k0 = 0; k0 < TOWER; k0 += 32) {
      const u16* ar = &Ts2[w * 16 + lo][k0 + hi * 8];
      U16bf a;
      a.q[0] = *(const U128*)(ar);
      a.q[1] = *(const U128*)(ar + 16);
      const u16* wr = wt8T + (size_t)lo * TOWER + k0 + hi * 8;
      U16bf b;
      b.q[0] = *(const U128*)(wr);
      b.q[1] = *(const U128*)(wr + 16);
      ce = wmma_bf16(a, b, ce);
    }
    if (lo < 8) {
      float bj = b_t8[lo];
      #pragma unroll
      for (int v = 0; v < 8; ++v)
        E8s[w * 16 + hi * 8 + v][lo] = ce[v] + bj;
    }
  }
  __syncthreads();

  // ------- RMS-norm + residual E8 quantization + analytic decode-LN stats -------
  if (t < MTILE) {
    float e[8];
    #pragma unroll
    for (int i = 0; i < 8; ++i) e[i] = E8s[t][i];
    float ms = 0.f;
    #pragma unroll
    for (int i = 0; i < 8; ++i) ms += e[i] * e[i];
    float rms = sqrtf(ms * (1.0f / 8.0f) + 1e-6f);
    float r_[8], q[8];
    #pragma unroll
    for (int i = 0; i < 8; ++i) { r_[i] = e[i] / rms * rms_g[i]; q[i] = 0.f; }
    for (int l = 0; l < NLEV; ++l) {
      float a[8], xs[8], bb[8];
      nd8(r_, a);
      #pragma unroll
      for (int i = 0; i < 8; ++i) xs[i] = r_[i] - 0.5f;
      nd8(xs, bb);
      #pragma unroll
      for (int i = 0; i < 8; ++i) bb[i] += 0.5f;
      float da = 0.f, db = 0.f;
      #pragma unroll
      for (int i = 0; i < 8; ++i) {
        float ea = r_[i] - a[i];  da += ea * ea;
        float eb = r_[i] - bb[i]; db += eb * eb;
      }
      #pragma unroll
      for (int i = 0; i < 8; ++i) {
        float ql = (da <= db) ? a[i] : bb[i];
        q[i] += ql; r_[i] -= ql;
      }
    }
    #pragma unroll
    for (int i = 0; i < 8; ++i) Qs[t][i] = q[i];
    // analytic mean/var of q @ W_8t + b_8t over 512 cols
    float mu = sb[80], ex2 = sb[81];
    #pragma unroll
    for (int j = 0; j < 8; ++j) mu  += q[j] * sb[j];
    #pragma unroll
    for (int j = 0; j < 8; ++j) ex2 += 2.0f * q[j] * sb[8 + j];
    #pragma unroll
    for (int j = 0; j < 8; ++j)
      #pragma unroll
      for (int k = 0; k < 8; ++k) ex2 += q[j] * q[k] * sb[16 + j * 8 + k];
    mu  *= (1.0f / 512.0f);
    ex2 *= (1.0f / 512.0f);
    float var = ex2 - mu * mu;
    RowMV[t][0] = mu;
    RowMV[t][1] = rsqrtf(var + 1e-6f);
  }
  __syncthreads();

  // -------- GEMM3 + decode-tower LN (analytic stats), write t2 bf16 --------
  {
    int r = t >> 2, nsegb = (t & 3) * 128;
    float qv[8];
    #pragma unroll
    for (int j = 0; j < 8; ++j) qv[j] = Qs[r][j];
    float mu = RowMV[r][0], rstd = RowMV[r][1];
    u16* orow = t2g + (size_t)(row0 + r) * TOWER + nsegb;
    for (int c0 = 0; c0 < 128; c0 += 8) {
      float y[8];
      #pragma unroll
      for (int c = 0; c < 8; ++c) {
        int n = nsegb + c0 + c;
        float acc = Tb_8t[n];
        #pragma unroll
        for (int j = 0; j < 8; ++j) acc += qv[j] * W_8t[j * TOWER + n];
        y[c] = (acc - mu) * rstd * Tg_dt[n] + Tb_dt[n];
      }
      U128 pkt;
      pkt.x = pk2(y[0], y[1]); pkt.y = pk2(y[2], y[3]);
      pkt.z = pk2(y[4], y[5]); pkt.w = pk2(y[6], y[7]);
      *(U128*)(orow + c0) = pkt;
    }
  }
}

// ================= decode: t2(bf16) @ W_tb + final LN over 2048 =================
__global__ __launch_bounds__(256) void e8_decode(
    const u16*  __restrict__ t2g,
    const u16*  __restrict__ wtbT,
    const float* __restrict__ b_tb,
    const float* __restrict__ ln_db_g, const float* __restrict__ ln_db_b,
    float* out) {

  __shared__ float RowStats[2 * MTILE];

  const int t  = threadIdx.x;
  const int w  = t >> 5;
  const int L  = t & 31;
  const int lo = L & 15;
  const int hi = L >> 4;
  const int mg = w >> 1;
  const int ng = w & 1;
  const int row0 = blockIdx.x * MTILE;

  if (t < 2 * MTILE) RowStats[t] = 0.0f;
  __syncthreads();

  const u16* ar0 = t2g + (size_t)(row0 + mg * 16 + lo) * TOWER + hi * 8;

  for (int nc = 0; nc < 4; ++nc) {
    v8f c4[16] = {};
    const int nbase = nc * 512 + ng * 256;
    for (int k0 = 0; k0 < TOWER; k0 += 32) {
      U16bf a;
      a.q[0] = *(const U128*)(ar0 + k0);
      a.q[1] = *(const U128*)(ar0 + k0 + 16);
      #pragma unroll
      for (int f = 0; f < 16; ++f) {
        int n = nbase + f * 16 + lo;
        const u16* wr = wtbT + (size_t)n * TOWER + k0 + hi * 8;
        U16bf b;
        b.q[0] = *(const U128*)(wr);
        b.q[1] = *(const U128*)(wr + 16);
        c4[f] = wmma_bf16(a, b, c4[f]);
      }
    }
    // bias + row-stat partials + stage pre-LN values to out
    float s[8], s2[8];
    #pragma unroll
    for (int v = 0; v < 8; ++v) { s[v] = 0.f; s2[v] = 0.f; }
    #pragma unroll
    for (int f = 0; f < 16; ++f) {
      float bias = b_tb[nbase + f * 16 + lo];
      #pragma unroll
      for (int v = 0; v < 8; ++v) {
        float z = c4[f][v] + bias;
        c4[f][v] = z; s[v] += z; s2[v] += z * z;
      }
    }
    #pragma unroll
    for (int m = 1; m < 16; m <<= 1) {
      #pragma unroll
      for (int v = 0; v < 8; ++v) {
        s[v]  += __shfl_xor(s[v],  m, 32);
        s2[v] += __shfl_xor(s2[v], m, 32);
      }
    }
    if (lo == 0) {
      #pragma unroll
      for (int v = 0; v < 8; ++v) {
        int rl = mg * 16 + hi * 8 + v;
        atomicAdd(&RowStats[rl * 2 + 0], s[v]);
        atomicAdd(&RowStats[rl * 2 + 1], s2[v]);
      }
    }
    #pragma unroll
    for (int f = 0; f < 16; ++f) {
      int col = nbase + f * 16 + lo;
      #pragma unroll
      for (int v = 0; v < 8; ++v)
        out[(size_t)(row0 + mg * 16 + hi * 8 + v) * BULK + col] = c4[f][v];
    }
  }

  __threadfence_block();
  __syncthreads();

  // coalesced normalize pass over the freshly written (L2-resident) tile
  for (int i = 0; i < 128; ++i) {
    int F   = i * 256 + t;          // float4 index within 64x2048 tile
    int r   = F >> 9;
    int c4i = (F & 511) * 4;
    float mu   = RowStats[r * 2 + 0] * (1.0f / 2048.0f);
    float var  = RowStats[r * 2 + 1] * (1.0f / 2048.0f) - mu * mu;
    float rstd = rsqrtf(var + 1e-6f);
    float* p = out + (size_t)(row0 + r) * BULK + c4i;
    float4 vv = *(const float4*)p;
    float4 g  = *(const float4*)(ln_db_g + c4i);
    float4 b  = *(const float4*)(ln_db_b + c4i);
    vv.x = (vv.x - mu) * rstd * g.x + b.x;
    vv.y = (vv.y - mu) * rstd * g.y + b.y;
    vv.z = (vv.z - mu) * rstd * g.z + b.z;
    vv.w = (vv.w - mu) * rstd * g.w + b.w;
    *(float4*)p = vv;
  }
}

extern "C" void kernel_launch(void* const* d_in, const int* in_sizes, int n_in,
                              void* d_out, int out_size, void* d_ws, size_t ws_size,
                              hipStream_t stream) {
  const float* x       = (const float*)d_in[0];
  const float* W_bt    = (const float*)d_in[1];
  const float* b_bt    = (const float*)d_in[2];
  const float* ln_et_g = (const float*)d_in[3];
  const float* ln_et_b = (const float*)d_in[4];
  const float* W_t8    = (const float*)d_in[5];
  const float* b_t8    = (const float*)d_in[6];
  const float* rms_g   = (const float*)d_in[7];
  const float* W_8t    = (const float*)d_in[8];
  const float* b_8t    = (const float*)d_in[9];
  const float* ln_dt_g = (const float*)d_in[10];
  const float* ln_dt_b = (const float*)d_in[11];
  const float* W_tb    = (const float*)d_in[12];
  const float* b_tb    = (const float*)d_in[13];
  const float* ln_db_g = (const float*)d_in[14];
  const float* ln_db_b = (const float*)d_in[15];
  float* out = (float*)d_out;

  u16* wbtT = (u16*)d_ws;                         // [512][2048] bf16, 2 MB
  u16* wtbT = wbtT + (size_t)BULK * TOWER;        // [2048][512] bf16, 2 MB
  u16* t2g  = wtbT + (size_t)BULK * TOWER;        // [16384][512] bf16, 16 MB
  u16* wt8T = t2g  + (size_t)16384 * TOWER;       // [16][512] bf16, 16 KB (at 20 MB)
  float* sb = (float*)((char*)d_ws + (size_t)20 * 1024 * 1024 + 16384);  // 82 floats

  const int prep_elems = 2 * BULK * TOWER + 16 * TOWER;
  e8_prep<<<(prep_elems + 255) / 256, 256, 0, stream>>>(W_bt, W_tb, W_t8, wbtT, wtbT, wt8T);
  e8_stats<<<1, 128, 0, stream>>>(W_8t, b_8t, sb);

  const int rows = 8 * 2048;
  e8_encode<<<rows / MTILE, 256, 0, stream>>>(
      x, wbtT, wt8T, b_bt, ln_et_g, ln_et_b, b_t8, rms_g,
      W_8t, b_8t, ln_dt_g, ln_dt_b, sb, t2g);
  e8_decode<<<rows / MTILE, 256, 0, stream>>>(
      t2g, wtbT, b_tb, ln_db_g, ln_db_b, out);
}
